// ToMeBlock_39951785788041
// MI455X (gfx1250) — matrix operand order repaired and checked by reading, not compiled, not run on hardware
//
#include <hip/hip_runtime.h>
#include <hip/hip_bf16.h>
#include <math.h>

// ---------------- problem constants (shape-specialized) ----------------
#define B_     4
#define N_     1568
#define C_     768
#define H_     12
#define D_     64
#define MLPH_  3072
#define R_     256
#define N2_    784           // N/2
#define NU_    528           // N2 - R
#define NOUT_  1312          // NU + N2
#define M1_    6272          // B*N
#define M2_    5248          // B*NOUT

typedef __attribute__((ext_vector_type(16))) _Float16 v16h;
typedef __attribute__((ext_vector_type(8)))  _Float16 v8h;
typedef __attribute__((ext_vector_type(8)))  float    v8f;

// ---------------- helpers ----------------
__device__ inline v16h cat8(v8h a, v8h b) {
  v16h r;
#pragma unroll
  for (int i = 0; i < 8; ++i) { r[i] = a[i]; r[8 + i] = b[i]; }
  return r;
}

__device__ inline v8f wmma16(v16h a, v16h b, v8f c) {
  // v_wmma_f32_16x16x32_f16 (8-arg: neg_a, A, neg_b, B, c_mod, C, reuse_a, reuse_b)
  return __builtin_amdgcn_wmma_f32_16x16x32_f16(false, a, false, b, (short)0, c, false, false);
}

__device__ inline float rmax16(float v) {
#pragma unroll
  for (int m = 1; m < 16; m <<= 1) v = fmaxf(v, __shfl_xor(v, m, 32));
  return v;
}
__device__ inline float rsum16(float v) {
#pragma unroll
  for (int m = 1; m < 16; m <<= 1) v += __shfl_xor(v, m, 32);
  return v;
}

// ---------------- elementwise prep kernels ----------------
__global__ void k_f32_to_f16(const float* __restrict__ s, _Float16* __restrict__ d, long n) {
  long i = (long)blockIdx.x * 256 + threadIdx.x;
  if (i < n) d[i] = (_Float16)s[i];
}

__global__ void k_log_size(const float* __restrict__ s, float* __restrict__ d, long n) {
  long i = (long)blockIdx.x * 256 + threadIdx.x;
  if (i < n) d[i] = logf(s[i]);
}

// ---------------- LayerNorm -> f16 ----------------
__global__ void k_ln_f16(const float* __restrict__ x, const float* __restrict__ g,
                         const float* __restrict__ bb, _Float16* __restrict__ h, int Cc) {
  const int row = blockIdx.x;
  const int tid = threadIdx.x;
  __shared__ float red[256];
  const float* xr = x + (long)row * Cc;
  float s = 0.f;
  for (int c = tid; c < Cc; c += 256) s += xr[c];
  red[tid] = s; __syncthreads();
  for (int st = 128; st > 0; st >>= 1) { if (tid < st) red[tid] += red[tid + st]; __syncthreads(); }
  float mu = red[0] / (float)Cc;
  __syncthreads();
  float s2 = 0.f;
  for (int c = tid; c < Cc; c += 256) { float d = xr[c] - mu; s2 += d * d; }
  red[tid] = s2; __syncthreads();
  for (int st = 128; st > 0; st >>= 1) { if (tid < st) red[tid] += red[tid + st]; __syncthreads(); }
  float rstd = rsqrtf(red[0] / (float)Cc + 1e-5f);
  for (int c = tid; c < Cc; c += 256)
    h[(long)row * Cc + c] = (_Float16)((xr[c] - mu) * rstd * g[c] + bb[c]);
}

// ---------------- epilogue selector ----------------
#define EP_F32      0
#define EP_F16      1
#define EP_GELU_F16 2
#define EP_RES_F32  3

template <int EP>
__device__ inline void ep_store(void* out, const float* res, long oidx, float v) {
  if (EP == EP_F32) {
    ((float*)out)[oidx] = v;
  } else if (EP == EP_F16) {
    ((_Float16*)out)[oidx] = (_Float16)v;
  } else if (EP == EP_GELU_F16) {
    float g = 0.5f * v * (1.f + erff(v * 0.70710678118f));   // exact GELU
    ((_Float16*)out)[oidx] = (_Float16)g;
  } else {
    ((float*)out)[oidx] = v + res[oidx];
  }
}

// ---------------- 1x1-tile WMMA GEMM (batched; used for scores) ----------------
// out = A(MxK) * Wt(NnxK)^T; fragments per the gfx1250 f16 WMMA VGPR layouts.
template <int EP>
__global__ void k_gemm_wmma(const _Float16* __restrict__ A, const _Float16* __restrict__ Wt,
                            const float* __restrict__ bias, const float* __restrict__ res,
                            void* __restrict__ out, int M, int Nn, int K,
                            long sA, long sB, long sO) {
  const int lane = threadIdx.x & 31;
  const int wave = threadIdx.x >> 5;
  const int tilesN = Nn >> 4;
  const int totalTiles = (M >> 4) * tilesN;
  const int tile = blockIdx.x * (blockDim.x >> 5) + wave;
  if (tile >= totalTiles) return;              // wave-uniform: EXEC stays all-ones for WMMA
  const int tm = tile / tilesN, tn = tile % tilesN;

  const _Float16* Ab = A + (long)blockIdx.y * sA;
  const _Float16* Bb = Wt + (long)blockIdx.y * sB;

  const int l15 = lane & 15;
  const int klo = (lane < 16) ? 0 : 8;
  const int kb  = (lane < 16) ? 0 : 16;
  const long arow = (long)(tm * 16 + l15) * K;
  const long brow = (long)(tn * 16 + l15) * K;

  v8f acc = {};
  for (int k0 = 0; k0 < K; k0 += 32) {
    v16h af = cat8(*(const v8h*)(Ab + arow + k0 + klo), *(const v8h*)(Ab + arow + k0 + 16 + klo));
    v16h bf = cat8(*(const v8h*)(Bb + brow + k0 + klo),  *(const v8h*)(Bb + brow + k0 + 16 + klo));
    acc = wmma16(af, bf, acc);
  }

  const int colo = tn * 16 + l15;
  const int rowb = tm * 16 + ((lane >> 4) << 3);
  const float bv = bias ? bias[colo] : 0.f;
#pragma unroll
  for (int j = 0; j < 8; ++j)
    ep_store<EP>(out, res, (long)blockIdx.y * sO + (long)(rowb + j) * Nn + colo, acc[j] + bv);
}

// ---------------- 2x2-tile WMMA GEMM (M,Nn multiples of 32; the heavy GEMMs) ----------------
// Each A/B fragment pair feeds 4 WMMAs -> 2x the WMMA-per-byte of the 1x1 kernel.
template <int EP>
__global__ void k_gemm_wmma2(const _Float16* __restrict__ A, const _Float16* __restrict__ Wt,
                             const float* __restrict__ bias, const float* __restrict__ res,
                             void* __restrict__ out, int M, int Nn, int K) {
  const int lane = threadIdx.x & 31;
  const int wave = threadIdx.x >> 5;
  const int tN2 = Nn >> 5;
  const int total = (M >> 5) * tN2;
  const int tile = blockIdx.x * (blockDim.x >> 5) + wave;
  if (tile >= total) return;
  const int tm = tile / tN2, tn = tile % tN2;

  const int l15 = lane & 15;
  const int klo = (lane < 16) ? 0 : 8;
  const int kb  = (lane < 16) ? 0 : 16;
  const long ar0 = (long)(tm * 32 + l15) * K;
  const long ar1 = ar0 + (long)16 * K;
  const long br0 = (long)(tn * 32 + l15) * K;
  const long br1 = br0 + (long)16 * K;

  v8f a00 = {}, a01 = {}, a10 = {}, a11 = {};
  for (int k0 = 0; k0 < K; k0 += 32) {
    v16h fa0 = cat8(*(const v8h*)(A + ar0 + k0 + klo), *(const v8h*)(A + ar0 + k0 + 16 + klo));
    v16h fa1 = cat8(*(const v8h*)(A + ar1 + k0 + klo), *(const v8h*)(A + ar1 + k0 + 16 + klo));
    v16h fb0 = cat8(*(const v8h*)(Wt + br0 + k0 + klo), *(const v8h*)(Wt + br0 + k0 + 16 + klo));
    v16h fb1 = cat8(*(const v8h*)(Wt + br1 + k0 + klo), *(const v8h*)(Wt + br1 + k0 + 16 + klo));
    if (k0 + 32 < K) {
      __builtin_prefetch(A + ar0 + k0 + 32, 0, 1);
      __builtin_prefetch(Wt + br0 + k0 + 32, 0, 1);
    }
    a00 = wmma16(fa0, fb0, a00);
    a01 = wmma16(fa0, fb1, a01);
    a10 = wmma16(fa1, fb0, a10);
    a11 = wmma16(fa1, fb1, a11);
  }

  const int rowb = tm * 32 + ((lane >> 4) << 3);
  const int c0 = tn * 32 + l15, c1 = c0 + 16;
  const float bv0 = bias ? bias[c0] : 0.f;
  const float bv1 = bias ? bias[c1] : 0.f;
#pragma unroll
  for (int j = 0; j < 8; ++j) {
    ep_store<EP>(out, res, (long)(rowb + j) * Nn + c0,      a00[j] + bv0);
    ep_store<EP>(out, res, (long)(rowb + j) * Nn + c1,      a01[j] + bv1);
    ep_store<EP>(out, res, (long)(rowb + 16 + j) * Nn + c0, a10[j] + bv0);
    ep_store<EP>(out, res, (long)(rowb + 16 + j) * Nn + c1, a11[j] + bv1);
  }
}

// ---------------- QKV repack: rowmajor [M1,3C] f16 -> q/k/v [b,h,n,d] f16 (+bias) ----------------
__global__ void k_repack_qkv(const _Float16* __restrict__ qkv, const float* __restrict__ q_bias,
                             const float* __restrict__ v_bias, _Float16* __restrict__ q16,
                             _Float16* __restrict__ k16, _Float16* __restrict__ v16) {
  long i = (long)blockIdx.x * 256 + threadIdx.x;
  if (i >= (long)B_ * H_ * N_ * D_) return;
  int d = (int)(i & (D_ - 1));
  long t = i >> 6;
  int n = (int)(t % N_); t /= N_;
  int hh = (int)(t % H_);
  int b = (int)(t / H_);
  const long rowoff = (long)(b * N_ + n) * (3 * C_);
  const int cq = hh * D_ + d;
  float qv = (float)qkv[rowoff + cq] + q_bias[cq];
  float kv = (float)qkv[rowoff + C_ + cq];
  float vv = (float)qkv[rowoff + 2 * C_ + cq] + v_bias[cq];
  const long oo = ((long)(b * H_ + hh) * N_ + n) * D_ + d;
  q16[oo] = (_Float16)qv; k16[oo] = (_Float16)kv; v16[oo] = (_Float16)vv;
}

// ---------------- flash attention: 2 waves/block, shared V^T staging ----------------
// block = 64 threads; grid = (N/32 q-tile pairs, H, B); lsz = precomputed log(size).
__global__ void k_attn_flash(const _Float16* __restrict__ q16, const _Float16* __restrict__ k16,
                             const _Float16* __restrict__ v16, const float* __restrict__ lsz,
                             _Float16* __restrict__ o16) {
  const int tid = threadIdx.x;
  const int lane = tid & 31;
  const int wave = tid >> 5;
  const int qt = blockIdx.x * 2 + wave;
  const int hh = blockIdx.y, b = blockIdx.z;
  const long bh = (long)(b * H_ + hh) * N_;

  __shared__ alignas(16) _Float16 vT[64 * 32];        // shared across both waves
  __shared__ alignas(16) _Float16 Pld[2][16 * 32];    // per-wave P staging

  const int l15 = lane & 15;
  const int hiHalf = lane >> 4;
  const int klo = hiHalf ? 8 : 0;
  const int kb  = hiHalf ? 16 : 0;

  // Q A-fragments for K-dim d = 0..31 and 32..63
  const _Float16* qp = q16 + (bh + qt * 16 + l15) * D_;
  v16h qa0 = cat8(*(const v8h*)(qp + klo),      *(const v8h*)(qp + 16 + klo));
  v16h qa1 = cat8(*(const v8h*)(qp + 32 + klo), *(const v8h*)(qp + 48 + klo));

  v8f oacc[4] = {{}, {}, {}, {}};
  float mrow[8], lrow[8];
#pragma unroll
  for (int j = 0; j < 8; ++j) { mrow[j] = -3.0e38f; lrow[j] = 0.f; }

  for (int kc = 0; kc < N_ / 32; ++kc) {
    const int key0 = kc * 32;

    // cooperative V^T staging (64 dims x 32 keys); fence reads of previous chunk first
    __syncthreads();
    for (int t = tid; t < 32 * 64; t += 64) {
      int key = t >> 6, d = t & 63;
      vT[d * 32 + key] = v16[(bh + key0 + key) * D_ + d];
    }
    __syncthreads();

    // S = scale * Q K^T + log(size): two 16x16 tiles (keys 0..15 and 16..31 of chunk)
    v8f s0 = {}, s1 = {};
    {
      const _Float16* kp = k16 + (bh + key0 + l15) * D_;
      s0 = wmma16(qa0, cat8(*(const v8h*)(kp + klo), *(const v8h*)(kp + 16 + klo)), s0);
      s0 = wmma16(qa1, cat8(*(const v8h*)(kp + 32 + klo), *(const v8h*)(kp + 48 + klo)), s0);
    }
    {
      const _Float16* kp = k16 + (bh + key0 + 16 + l15) * D_;
      s1 = wmma16(qa0, cat8(*(const v8h*)(kp + klo), *(const v8h*)(kp + 16 + klo)), s1);
      s1 = wmma16(qa1, cat8(*(const v8h*)(kp + 32 + klo), *(const v8h*)(kp + 48 + klo)), s1);
    }
    const float lb0 = lsz[(long)b * N_ + key0 + l15];
    const float lb1 = lsz[(long)b * N_ + key0 + 16 + l15];

    // online softmax (rows j + 8*hiHalf live across the 16 lanes of each half-wave)
#pragma unroll
    for (int j = 0; j < 8; ++j) {
      float a0 = s0[j] * 0.125f + lb0;     // scale = D^-0.5
      float a1 = s1[j] * 0.125f + lb1;
      float rm = rmax16(fmaxf(a0, a1));
      float mnew = fmaxf(mrow[j], rm);
      float alpha = expf(mrow[j] - mnew);
      float p0 = expf(a0 - mnew), p1 = expf(a1 - mnew);
      lrow[j] = lrow[j] * alpha + rsum16(p0 + p1);
      mrow[j] = mnew;
      oacc[0][j] *= alpha; oacc[1][j] *= alpha; oacc[2][j] *= alpha; oacc[3][j] *= alpha;
      const int prow = j + (hiHalf << 3);
      Pld[wave][prow * 32 + l15]      = (_Float16)p0;   // per-wave buffer: DS in-order, no barrier
      Pld[wave][prow * 32 + 16 + l15] = (_Float16)p1;
    }

    // P (16x32) @ V (32x64)
    const _Float16* pp = &Pld[wave][l15 * 32];
    v16h pa = cat8(*(const v8h*)(pp + klo), *(const v8h*)(pp + 16 + klo));
#pragma unroll
    for (int t = 0; t < 4; ++t) {
      const _Float16* vp = &vT[(t * 16 + l15) * 32 + klo];
      oacc[t] = wmma16(pa, cat8(*(const v8h*)(vp), *(const v8h*)(vp + 16)), oacc[t]);
    }
  }

  // epilogue: o = O / l, write to [b, n, h*64 + d] f16
#pragma unroll
  for (int j = 0; j < 8; ++j) {
    const int row = qt * 16 + j + (hiHalf << 3);
    const float rl = 1.f / lrow[j];
    const long ob = ((long)(b * N_ + row)) * C_ + hh * D_ + l15;
    o16[ob]      = (_Float16)(oacc[0][j] * rl);
    o16[ob + 16] = (_Float16)(oacc[1][j] * rl);
    o16[ob + 32] = (_Float16)(oacc[2][j] * rl);
    o16[ob + 48] = (_Float16)(oacc[3][j] * rl);
  }
}

// ---------------- metric = mean_h(k), normalized; split into a (even) / bt (odd) ----------------
__global__ void k_metric_norm(const _Float16* __restrict__ k16, _Float16* __restrict__ a16,
                              _Float16* __restrict__ bt16) {
  const int bn = blockIdx.x;           // b*N + n
  const int d = threadIdx.x;           // 64 threads
  const int b = bn / N_, n = bn % N_;
  float s = 0.f;
  for (int hh = 0; hh < H_; ++hh)
    s += (float)k16[((long)(b * H_ + hh) * N_ + n) * D_ + d];
  s *= (1.f / H_);
  __shared__ float red[64];
  red[d] = s * s; __syncthreads();
  for (int st = 32; st > 0; st >>= 1) { if (d < st) red[d] += red[d + st]; __syncthreads(); }
  const float rn = rsqrtf(red[0]);
  const _Float16 mv = (_Float16)(s * rn);
  if ((n & 1) == 0) a16[((long)(b * N2_ + (n >> 1))) * D_ + d] = mv;
  else              bt16[((long)(b * N2_ + (n >> 1))) * D_ + d] = mv;
}

// ---------------- per-row max + argmax over scores ----------------
__global__ void k_row_argmax(const float* __restrict__ sc, float* __restrict__ nmax,
                             int* __restrict__ nidx) {
  const int row = blockIdx.x;          // b*N2 + i
  const int tid = threadIdx.x;         // 128
  const float* r = sc + (long)row * N2_;
  float bv = -3.4e38f; int bi = 0;
  for (int j = tid; j < N2_; j += 128) {
    float v = r[j];
    if (v > bv) { bv = v; bi = j; }
  }
  __shared__ float rv[128]; __shared__ int ri[128];
  rv[tid] = bv; ri[tid] = bi; __syncthreads();
  for (int st = 64; st > 0; st >>= 1) {
    if (tid < st) {
      float v2 = rv[tid + st]; int i2 = ri[tid + st];
      if (v2 > rv[tid] || (v2 == rv[tid] && i2 < ri[tid])) { rv[tid] = v2; ri[tid] = i2; }
    }
    __syncthreads();
  }
  if (tid == 0) { nmax[row] = rv[0]; nidx[row] = ri[0]; }
}

// ---------------- bitonic argsort of -node_max (ascending, stable via index tie-break) ----------
__global__ void k_sort_select(const float* __restrict__ nmax, const int* __restrict__ nidx,
                              int* __restrict__ unm, int* __restrict__ srcI,
                              int* __restrict__ dstI) {
  const int b = blockIdx.x;
  const int tid = threadIdx.x;         // 1024
  __shared__ float sv[1024]; __shared__ int si[1024];
  if (tid < N2_) { sv[tid] = -nmax[b * N2_ + tid]; si[tid] = tid; }
  else           { sv[tid] = 3.4e38f;              si[tid] = tid; }
  __syncthreads();
  for (int ksz = 2; ksz <= 1024; ksz <<= 1)
    for (int jsz = ksz >> 1; jsz > 0; jsz >>= 1) {
      const int ixj = tid ^ jsz;
      if (ixj > tid) {
        float v1 = sv[tid], v2 = sv[ixj]; int i1 = si[tid], i2 = si[ixj];
        bool swapAsc = (v1 > v2) || (v1 == v2 && i1 > i2);
        bool doSwap = ((tid & ksz) == 0) ? swapAsc : !swapAsc;
        if (doSwap) { sv[tid] = v2; sv[ixj] = v1; si[tid] = i2; si[ixj] = i1; }
      }
      __syncthreads();
    }
  if (tid < R_) {
    const int s = si[tid];
    srcI[b * R_ + tid] = s;
    dstI[b * R_ + tid] = nidx[b * N2_ + s];
  } else if (tid < N2_) {
    unm[b * NU_ + (tid - R_)] = si[tid];
  }
}

// ---------------- token merge (deterministic: per-dst scan of the R matches) + divide -----------
__global__ void k_merge_tokens(const float* __restrict__ x1, const float* __restrict__ size,
                               const int* __restrict__ unm, const int* __restrict__ srcI,
                               const int* __restrict__ dstI, float* __restrict__ xm) {
  const int b = blockIdx.y;
  const long idx = (long)blockIdx.x * 256 + threadIdx.x;
  if (idx >= (long)NOUT_ * C_) return;
  const int j = (int)(idx / C_), c = (int)(idx % C_);
  float acc, sz;
  if (j < NU_) {
    const int tok = 2 * unm[b * NU_ + j];                 // unmerged src token
    sz = size[(long)b * N_ + tok];
    acc = x1[((long)b * N_ + tok) * C_ + c] * sz;
  } else {
    const int dj = j - NU_;
    const int tok = 2 * dj + 1;                           // dst token
    sz = size[(long)b * N_ + tok];
    acc = x1[((long)b * N_ + tok) * C_ + c] * sz;
    for (int s = 0; s < R_; ++s) {
      if (dstI[b * R_ + s] == dj) {
        const int st = 2 * srcI[b * R_ + s];
        const float ss = size[(long)b * N_ + st];
        acc += x1[((long)b * N_ + st) * C_ + c] * ss;
        sz += ss;
      }
    }
  }
  xm[((long)b * NOUT_ + j) * C_ + c] = acc / sz;          // x = merged / size_new
}

// ---------------- host-side orchestration ----------------
extern "C" void kernel_launch(void* const* d_in, const int* in_sizes, int n_in,
                              void* d_out, int out_size, void* d_ws, size_t ws_size,
                              hipStream_t stream) {
  (void)in_sizes; (void)n_in; (void)out_size; (void)ws_size;
  const float* x      = (const float*)d_in[0];
  const float* size   = (const float*)d_in[1];
  const float* qkv_w  = (const float*)d_in[2];
  const float* q_bias = (const float*)d_in[3];
  const float* v_bias = (const float*)d_in[4];
  const float* proj_w = (const float*)d_in[5];
  const float* proj_b = (const float*)d_in[6];
  const float* ln1_g  = (const float*)d_in[7];
  const float* ln1_b  = (const float*)d_in[8];
  const float* ln2_g  = (const float*)d_in[9];
  const float* ln2_b  = (const float*)d_in[10];
  const float* fc1_w  = (const float*)d_in[11];
  const float* fc1_b  = (const float*)d_in[12];
  const float* fc2_w  = (const float*)d_in[13];
  const float* fc2_b  = (const float*)d_in[14];

  char* w = (char*)d_ws;
  size_t off = 0;
  auto alloc = [&](size_t bytes) -> char* {
    char* p = w + off; off = (off + bytes + 255) & ~(size_t)255; return p;
  };

  _Float16* h16     = (_Float16*)alloc((size_t)M1_ * C_ * 2);
  _Float16* qkvw16  = (_Float16*)alloc((size_t)3 * C_ * C_ * 2);
  _Float16* projw16 = (_Float16*)alloc((size_t)C_ * C_ * 2);
  _Float16* fc1w16  = (_Float16*)alloc((size_t)MLPH_ * C_ * 2);
  _Float16* fc2w16  = (_Float16*)alloc((size_t)C_ * MLPH_ * 2);
  _Float16* qkv16   = (_Float16*)alloc((size_t)M1_ * 3 * C_ * 2);
  _Float16* q16     = (_Float16*)alloc((size_t)B_ * H_ * N_ * D_ * 2);
  _Float16* k16     = (_Float16*)alloc((size_t)B_ * H_ * N_ * D_ * 2);
  _Float16* v16     = (_Float16*)alloc((size_t)B_ * H_ * N_ * D_ * 2);
  float*    lsz     = (float*)alloc((size_t)B_ * N_ * 4);
  _Float16* o16     = (_Float16*)alloc((size_t)M1_ * C_ * 2);
  float*    x1      = (float*)alloc((size_t)M1_ * C_ * 4);
  _Float16* a16     = (_Float16*)alloc((size_t)B_ * N2_ * D_ * 2);
  _Float16* bt16    = (_Float16*)alloc((size_t)B_ * N2_ * D_ * 2);
  float*    scores  = (float*)alloc((size_t)B_ * N2_ * N2_ * 4);
  float*    nmax    = (float*)alloc((size_t)B_ * N2_ * 4);
  int*      nidx    = (int*)alloc((size_t)B_ * N2_ * 4);
  int*      unmI    = (int*)alloc((size_t)B_ * NU_ * 4);
  int*      srcI    = (int*)alloc((size_t)B_ * R_ * 4);
  int*      dstI    = (int*)alloc((size_t)B_ * R_ * 4);
  float*    xm      = (float*)alloc((size_t)B_ * NOUT_ * C_ * 4);
  _Float16* h2_16   = (_Float16*)alloc((size_t)B_ * NOUT_ * C_ * 2);
  _Float16* g16     = (_Float16*)alloc((size_t)B_ * NOUT_ * MLPH_ * 2);

  auto blks = [](long n) { return (int)((n + 255) / 256); };

  // 1) weights -> f16 ; log(size) precompute
  k_f32_to_f16<<<blks((long)3 * C_ * C_), 256, 0, stream>>>(qkv_w, qkvw16, (long)3 * C_ * C_);
  k_f32_to_f16<<<blks((long)C_ * C_),     256, 0, stream>>>(proj_w, projw16, (long)C_ * C_);
  k_f32_to_f16<<<blks((long)MLPH_ * C_),  256, 0, stream>>>(fc1_w, fc1w16, (long)MLPH_ * C_);
  k_f32_to_f16<<<blks((long)C_ * MLPH_),  256, 0, stream>>>(fc2_w, fc2w16, (long)C_ * MLPH_);
  k_log_size<<<blks((long)B_ * N_), 256, 0, stream>>>(size, lsz, (long)B_ * N_);

  // 2) LN1
  k_ln_f16<<<M1_, 256, 0, stream>>>(x, ln1_g, ln1_b, h16, C_);

  // 3) QKV GEMM (6272x768 @ 2304x768^T) -> f16
  {
    int tiles = (M1_ / 32) * (3 * C_ / 32);
    k_gemm_wmma2<EP_F16><<<dim3((tiles + 7) / 8), 256, 0, stream>>>(
        h16, qkvw16, nullptr, nullptr, qkv16, M1_, 3 * C_, C_);
  }

  // 4) repack + bias -> q/k/v [b,h,n,d]
  k_repack_qkv<<<blks((long)B_ * H_ * N_ * D_), 256, 0, stream>>>(qkv16, q_bias, v_bias, q16, k16, v16);

  // 5) flash attention -> o16 [b,n,C]   (2 q-tiles per block share V^T staging)
  k_attn_flash<<<dim3(N_ / 32, H_, B_), 64, 0, stream>>>(q16, k16, v16, lsz, o16);

  // 6) proj GEMM + residual: x1 = x + o @ proj_w^T + proj_b
  {
    int tiles = (M1_ / 32) * (C_ / 32);
    k_gemm_wmma2<EP_RES_F32><<<dim3((tiles + 7) / 8), 256, 0, stream>>>(
        o16, projw16, proj_b, x, x1, M1_, C_, C_);
  }

  // 7) metric -> normalized, split even/odd
  k_metric_norm<<<B_ * N_, 64, 0, stream>>>(k16, a16, bt16);

  // 8) scores = a @ bt^T (batched over B; 49x49 tiles -> 1x1 kernel)
  {
    int tiles = (N2_ / 16) * (N2_ / 16);
    k_gemm_wmma<EP_F32><<<dim3((tiles + 7) / 8, B_), 256, 0, stream>>>(
        a16, bt16, nullptr, nullptr, scores, N2_, N2_, D_,
        (long)N2_ * D_, (long)N2_ * D_, (long)N2_ * N2_);
  }

  // 9) node max / argmax
  k_row_argmax<<<B_ * N2_, 128, 0, stream>>>(scores, nmax, nidx);

  // 10) sort + select unm/src/dst indices
  k_sort_select<<<B_, 1024, 0, stream>>>(nmax, nidx, unmI, srcI, dstI);

  // 11) merge (weighted sum + divide) -> xm = merged x
  k_merge_tokens<<<dim3(blks((long)NOUT_ * C_), B_), 256, 0, stream>>>(x1, size, unmI, srcI, dstI, xm);

  // 12) LN2
  k_ln_f16<<<M2_, 256, 0, stream>>>(xm, ln2_g, ln2_b, h2_16, C_);

  // 13) fc1 GEMM + exact GELU -> f16
  {
    int tiles = (M2_ / 32) * (MLPH_ / 32);
    k_gemm_wmma2<EP_GELU_F16><<<dim3((tiles + 7) / 8), 256, 0, stream>>>(
        h2_16, fc1w16, fc1_b, nullptr, g16, M2_, MLPH_, C_);
  }

  // 14) fc2 GEMM + residual -> d_out (f32, B x 1312 x 768)
  {
    int tiles = (M2_ / 32) * (C_ / 32);
    k_gemm_wmma2<EP_RES_F32><<<dim3((tiles + 7) / 8), 256, 0, stream>>>(
        g16, fc2w16, fc2_b, xm, d_out, M2_, C_, MLPH_);
  }
}
